// CLUBForCategorical_2585570312816
// MI455X (gfx1250) — matrix operand order repaired and checked by reading, not compiled, not run on hardware
//
#include <hip/hip_runtime.h>

typedef __attribute__((ext_vector_type(16))) _Float16 v16h;
typedef __attribute__((ext_vector_type(8)))  _Float16 v8h;
typedef __attribute__((ext_vector_type(4)))  _Float16 v4h;
typedef __attribute__((ext_vector_type(8)))  float    v8f;
typedef __attribute__((ext_vector_type(4)))  float    v4f;
typedef __attribute__((ext_vector_type(4)))  int      v4i;
typedef __attribute__((ext_vector_type(8)))  int      v8i;

#define N_ROWS 16384
#define D_DIM  1024
#define L_CLS  1000
#define L_PAD  1024

#define BM 128
#define BN 128
#define BK 32
#define PAD_H 8              // halves of padding per LDS row (bank-spread)
#define LDA (BK + PAD_H)     // 40 halves = 80 bytes (16B-aligned rows)
#define LDB (BK + PAD_H)

#if __has_builtin(__builtin_amdgcn_s_wait_asynccnt)
#define WAIT_ASYNC(n) __builtin_amdgcn_s_wait_asynccnt((short)(n))
#else
#define WAIT_ASYNC(n) asm volatile("s_wait_asynccnt %0" :: "i"(n) : "memory")
#endif

#if __has_builtin(__builtin_amdgcn_s_wait_tensorcnt)
#define WAIT_TENSOR(n) __builtin_amdgcn_s_wait_tensorcnt((short)(n))
#else
#define WAIT_TENSOR(n) asm volatile("s_wait_tensorcnt %0" :: "i"(n) : "memory")
#endif

__device__ __forceinline__ unsigned lds_off(const void* p) {
  // Generic LDS pointers: low 32 bits are the LDS byte offset.
  return (unsigned)(size_t)p;
}

// ---------------------------------------------------------------------------
// TDM: issue one 2-D tile load (128 rows x 64B) global->LDS via the Tensor
// Data Mover. data_size=8B; hardware inserts 16B LDS padding after each 64B
// row so rows land on the 80B-pitch LDS layout (LDA*2).
// ---------------------------------------------------------------------------
__device__ __forceinline__ void tdm_load_tile(unsigned lds_addr, uint64_t gaddr) {
  v4i g0;
  g0[0] = 1;                                        // count=1, user descriptor
  g0[1] = (int)lds_addr;                            // lds_addr
  g0[2] = (int)(unsigned)(gaddr & 0xffffffffu);     // global_addr[31:0]
  g0[3] = (int)(((unsigned)(gaddr >> 32) & 0x01ffffffu) | 0x80000000u); // +type=2
  v8i g1;
  g1[0] = (int)((3u << 16)      // data_size = 8B
              | (1u << 20)      // pad_enable
              | (3u << 22)      // pad_interval: 16 DWORDs = 64B
              | (3u << 25));    // pad_amount:   4 DWORDs = 16B
  g1[1] = (int)(8u << 16);      // tensor_dim0 = 8 units (64B)
  g1[2] = (int)(128u << 16);    // tensor_dim1 = 128 rows
  g1[3] = (int)(8u << 16);      // tile_dim0 = 8 units
  g1[4] = 128;                  // tile_dim1 = 128 rows (tile_dim2 = 0)
  g1[5] = 256;                  // tensor_dim0_stride = 2048B / 8B
  g1[6] = 0;
  g1[7] = 0;
  v4i gz;
  gz[0] = 0; gz[1] = 0; gz[2] = 0; gz[3] = 0;       // groups 2/3: dims unused
  asm volatile("tensor_load_to_lds %0, %1, %2, %3"
               :: "s"(g0), "s"(g1), "s"(gz), "s"(gz) : "memory");
}

// ---------------------------------------------------------------------------
// Zero the workspace accumulators (colsum[L_PAD], diag[N_ROWS]).
// ---------------------------------------------------------------------------
__global__ void club_init_ws(float* __restrict__ colsum, float* __restrict__ diag) {
  int i = blockIdx.x * blockDim.x + threadIdx.x;
  if (i < L_PAD)  colsum[i] = 0.0f;
  if (i < N_ROWS) diag[i]   = 0.0f;
}

// ---------------------------------------------------------------------------
// One-time f32 -> f16 conversion passes (X16 [N,D], W16 [L_PAD,D] zero-padded)
// ---------------------------------------------------------------------------
__global__ __launch_bounds__(256)
void club_cvt_x(const float* __restrict__ X, _Float16* __restrict__ X16) {
  size_t i = (size_t)blockIdx.x * blockDim.x + threadIdx.x;   // float4 index
  v4f x = *(const v4f*)(X + i * 4);
  v4h h;
  h[0] = (_Float16)x[0]; h[1] = (_Float16)x[1];
  h[2] = (_Float16)x[2]; h[3] = (_Float16)x[3];
  *(v4h*)(X16 + i * 4) = h;
}

__global__ __launch_bounds__(256)
void club_cvt_w(const float* __restrict__ W, _Float16* __restrict__ W16) {
  size_t i = (size_t)blockIdx.x * blockDim.x + threadIdx.x;   // float4 index
  int row = (int)(i >> 8);                                    // D/4 chunks/row
  v4f x = (v4f)0.0f;
  if (row < L_CLS) x = *(const v4f*)(W + i * 4);
  v4h h;
  h[0] = (_Float16)x[0]; h[1] = (_Float16)x[1];
  h[2] = (_Float16)x[2]; h[3] = (_Float16)x[3];
  *(v4h*)(W16 + i * 4) = h;
}

// ---------------------------------------------------------------------------
// Shared GEMM body pieces (fragments + WMMA + epilogue) as macros so all three
// staging variants stay in sync.
// ---------------------------------------------------------------------------
#define CLUB_COMPUTE_STAGE(APTR, BPTR)                                        \
  {                                                                           \
    v16h afrag[2];                                                            \
    _Pragma("unroll")                                                         \
    for (int mi = 0; mi < 2; ++mi) {                                          \
      int arow = wave_m * 32 + mi * 16 + (lane & 15);                         \
      int ks   = (lane >> 4) * 8;                                             \
      v8h a0 = *(const v8h*)&(APTR)[arow * LDA + ks];                         \
      v8h a1 = *(const v8h*)&(APTR)[arow * LDA + ks + 16];                    \
      afrag[mi] = __builtin_shufflevector(a0, a1,                             \
          0,1,2,3,4,5,6,7,8,9,10,11,12,13,14,15);                             \
    }                                                                         \
    _Pragma("unroll")                                                         \
    for (int ni = 0; ni < 4; ++ni) {                                          \
      int bcol = wave_n * 64 + ni * 16 + (lane & 15);                         \
      int ks   = (lane >> 4) * 8;                                             \
      v8h b0 = *(const v8h*)&(BPTR)[bcol * LDB + ks];                         \
      v8h b1 = *(const v8h*)&(BPTR)[bcol * LDB + ks + 16];                    \
      v16h bfrag = __builtin_shufflevector(b0, b1,                            \
          0,1,2,3,4,5,6,7,8,9,10,11,12,13,14,15);                             \
      _Pragma("unroll")                                                       \
      for (int mi = 0; mi < 2; ++mi) {                                        \
        acc[mi][ni] = __builtin_amdgcn_wmma_f32_16x16x32_f16(                 \
            false, afrag[mi], false, bfrag, (short)0, acc[mi][ni],            \
            false, false);                                                    \
      }                                                                       \
    }                                                                         \
  }

#define CLUB_EPILOGUE()                                                       \
  {                                                                           \
    _Pragma("unroll")                                                         \
    for (int ni = 0; ni < 4; ++ni) {                                          \
      float s = 0.0f;                                                         \
      _Pragma("unroll")                                                       \
      for (int mi = 0; mi < 2; ++mi)                                          \
        _Pragma("unroll")                                                     \
        for (int j = 0; j < 8; ++j)                                           \
          s += acc[mi][ni][j];                                                \
      s += __shfl_xor(s, 16, 32);                                             \
      if (lane < 16)                                                          \
        atomicAdd(&colsum_lds[wave_n * 64 + ni * 16 + lane], s);              \
    }                                                                         \
    _Pragma("unroll")                                                         \
    for (int mi = 0; mi < 2; ++mi) {                                          \
      _Pragma("unroll")                                                       \
      for (int j = 0; j < 8; ++j) {                                           \
        int rlocal = wave_m * 32 + mi * 16 + (lane >> 4) * 8 + j;             \
        int tc = tcol[rlocal];                                                \
        _Pragma("unroll")                                                     \
        for (int ni = 0; ni < 4; ++ni) {                                      \
          int clocal = wave_n * 64 + ni * 16 + (lane & 15);                   \
          if (tc == clocal)                                                   \
            diag[m0 + rlocal] = acc[mi][ni][j];                               \
        }                                                                     \
      }                                                                       \
    }                                                                         \
    __syncthreads();                                                          \
    if (tid < BN) {                                                           \
      int l = l0 + tid;                                                       \
      if (l < L_CLS)                                                          \
        atomicAdd(&colsum[l], colsum_lds[tid]);                               \
    }                                                                         \
  }

// ---------------------------------------------------------------------------
// Variant 1 (primary): f16 inputs, TDM (tensor_load_to_lds) staging,
// double-buffered, wave 0 drives the DMA, TENSORcnt pipelining.
// ---------------------------------------------------------------------------
__global__ __launch_bounds__(256)
void club_gemm_f16_tdm(const _Float16* __restrict__ X16,   // [N_ROWS, D_DIM]
                       const _Float16* __restrict__ W16,   // [L_PAD, D_DIM]
                       const int*      __restrict__ labels,
                       float* __restrict__ colsum,         // [L_PAD]
                       float* __restrict__ diag)           // [N_ROWS]
{
  __shared__ __align__(16) _Float16 As[2][BM * LDA];
  __shared__ __align__(16) _Float16 Bs[2][BN * LDB];
  __shared__ float colsum_lds[BN];
  __shared__ int   tcol[BM];

  const int tid    = (int)threadIdx.x;
  const int lane   = tid & 31;
  const int wv     = tid >> 5;
  const int wave_m = wv & 3;
  const int wave_n = wv >> 2;

  const int m0 = (int)blockIdx.y * BM;
  const int l0 = (int)blockIdx.x * BN;

  if (tid < BN) colsum_lds[tid] = 0.0f;
  if (tid < BM) tcol[tid] = labels[m0 + tid] - l0;

  v8f acc[2][4];
#pragma unroll
  for (int mi = 0; mi < 2; ++mi)
#pragma unroll
    for (int ni = 0; ni < 4; ++ni)
      acc[mi][ni] = (v8f)0.0f;

  const unsigned ldsA[2] = { lds_off(&As[0][0]), lds_off(&As[1][0]) };
  const unsigned ldsB[2] = { lds_off(&Bs[0][0]), lds_off(&Bs[1][0]) };
  const uint64_t baseX = (uint64_t)(size_t)(X16 + (size_t)m0 * D_DIM);
  const uint64_t baseW = (uint64_t)(size_t)(W16 + (size_t)l0 * D_DIM);

  const int S = D_DIM / BK;   // 32 stages, 64B of K per stage
  if (wv == 0) {
    tdm_load_tile(ldsA[0], baseX);
    tdm_load_tile(ldsB[0], baseW);
    tdm_load_tile(ldsA[1], baseX + BK * 2);
    tdm_load_tile(ldsB[1], baseW + BK * 2);
  }

  for (int s = 0; s < S; ++s) {
    const int buf = s & 1;
    if (wv == 0) {
      if (s + 1 < S) { WAIT_TENSOR(2); } else { WAIT_TENSOR(0); }
    }
    __syncthreads();                    // stage s visible to all waves
    CLUB_COMPUTE_STAGE(As[buf], Bs[buf]);
    __syncthreads();                    // all reads of buf done
    if (s + 2 < S && wv == 0) {
      uint64_t kb = (uint64_t)(s + 2) * (BK * 2);
      tdm_load_tile(ldsA[buf], baseX + kb);
      tdm_load_tile(ldsB[buf], baseW + kb);
    }
  }

  CLUB_EPILOGUE();
}

// ---------------------------------------------------------------------------
// Variant 2: f16 inputs, per-lane async global->LDS staging.
// ---------------------------------------------------------------------------
__global__ __launch_bounds__(256)
void club_gemm_f16_async(const _Float16* __restrict__ X16,
                         const _Float16* __restrict__ W16,
                         const int*      __restrict__ labels,
                         float* __restrict__ colsum,
                         float* __restrict__ diag)
{
  __shared__ __align__(16) _Float16 As[2][BM * LDA];
  __shared__ __align__(16) _Float16 Bs[2][BN * LDB];
  __shared__ float colsum_lds[BN];
  __shared__ int   tcol[BM];

  const int tid    = (int)threadIdx.x;
  const int lane   = tid & 31;
  const int wv     = tid >> 5;
  const int wave_m = wv & 3;
  const int wave_n = wv >> 2;

  const int m0 = (int)blockIdx.y * BM;
  const int l0 = (int)blockIdx.x * BN;

  if (tid < BN) colsum_lds[tid] = 0.0f;
  if (tid < BM) tcol[tid] = labels[m0 + tid] - l0;

  v8f acc[2][4];
#pragma unroll
  for (int mi = 0; mi < 2; ++mi)
#pragma unroll
    for (int ni = 0; ni < 4; ++ni)
      acc[mi][ni] = (v8f)0.0f;

  const int crow   = tid >> 2;              // 0..63
  const int cbyte  = (tid & 3) * 16;        // 0/16/32/48
  const unsigned ldsRow  = (unsigned)(crow * (LDA * 2) + cbyte);
  const unsigned ldsRow2 = ldsRow + 64u * (LDA * 2);
  const unsigned gRow    = (unsigned)(crow * (D_DIM * 2) + cbyte);
  const unsigned gRow2   = gRow + 64u * (D_DIM * 2);
  const unsigned ldsA0 = lds_off(&As[0][0]);
  const unsigned ldsA1 = lds_off(&As[1][0]);
  const unsigned ldsB0 = lds_off(&Bs[0][0]);
  const unsigned ldsB1 = lds_off(&Bs[1][0]);
  const uint64_t baseX = (uint64_t)(size_t)(X16 + (size_t)m0 * D_DIM);
  const uint64_t baseW = (uint64_t)(size_t)(W16 + (size_t)l0 * D_DIM);

#define ISSUE_STAGE(s, ldsAb, ldsBb)                                          \
  {                                                                           \
    unsigned k2 = (unsigned)((s) * BK * 2);                                   \
    unsigned a0 = (ldsAb) + ldsRow, a1 = (ldsAb) + ldsRow2;                   \
    unsigned b0 = (ldsBb) + ldsRow, b1 = (ldsBb) + ldsRow2;                   \
    unsigned g0 = gRow + k2, g1 = gRow2 + k2;                                 \
    asm volatile(                                                             \
        "global_load_async_to_lds_b128 %0, %2, %4\n\t"                        \
        "global_load_async_to_lds_b128 %1, %3, %4"                            \
        :: "v"(a0), "v"(a1), "v"(g0), "v"(g1), "s"(baseX) : "memory");        \
    asm volatile(                                                             \
        "global_load_async_to_lds_b128 %0, %2, %4\n\t"                        \
        "global_load_async_to_lds_b128 %1, %3, %4"                            \
        :: "v"(b0), "v"(b1), "v"(g0), "v"(g1), "s"(baseW) : "memory");        \
  }

  const int S = D_DIM / BK;
  ISSUE_STAGE(0, ldsA0, ldsB0);
  ISSUE_STAGE(1, ldsA1, ldsB1);

  for (int s = 0; s < S; ++s) {
    const int buf = s & 1;
    if (s + 1 < S) { WAIT_ASYNC(4); } else { WAIT_ASYNC(0); }
    __syncthreads();
    CLUB_COMPUTE_STAGE(As[buf], Bs[buf]);
    __syncthreads();
    if (s + 2 < S) {
      if (buf == 0) { ISSUE_STAGE(s + 2, ldsA0, ldsB0); }
      else          { ISSUE_STAGE(s + 2, ldsA1, ldsB1); }
    }
  }
#undef ISSUE_STAGE

  CLUB_EPILOGUE();
}

// ---------------------------------------------------------------------------
// Variant 3 (fallback when ws too small): f32 inputs, VGPR-staged + in-loop
// conversion.
// ---------------------------------------------------------------------------
__global__ __launch_bounds__(256)
void club_gemm_f32(const float* __restrict__ X,
                   const float* __restrict__ W,
                   const int*   __restrict__ labels,
                   float* __restrict__ colsum,
                   float* __restrict__ diag)
{
  __shared__ _Float16 As[BM * LDA];
  __shared__ _Float16 Bs[BN * LDB];
  __shared__ float    colsum_lds[BN];
  __shared__ int      tcol[BM];

  const int tid    = (int)threadIdx.x;
  const int lane   = tid & 31;
  const int wv     = tid >> 5;
  const int wave_m = wv & 3;
  const int wave_n = wv >> 2;

  const int m0 = (int)blockIdx.y * BM;
  const int l0 = (int)blockIdx.x * BN;

  if (tid < BN) colsum_lds[tid] = 0.0f;
  if (tid < BM) tcol[tid] = labels[m0 + tid] - l0;

  v8f acc[2][4];
#pragma unroll
  for (int mi = 0; mi < 2; ++mi)
#pragma unroll
    for (int ni = 0; ni < 4; ++ni)
      acc[mi][ni] = (v8f)0.0f;

  const int lr = tid >> 3;
  const int lc = (tid & 7) * 4;

  for (int k0 = 0; k0 < D_DIM; k0 += BK) {
    __syncthreads();
#pragma unroll
    for (int i = 0; i < 4; ++i) {
      int r = lr + i * 32;
      v4f xa = *(const v4f*)(X + (size_t)(m0 + r) * D_DIM + k0 + lc);
      v4h ha;
      ha[0] = (_Float16)xa[0]; ha[1] = (_Float16)xa[1];
      ha[2] = (_Float16)xa[2]; ha[3] = (_Float16)xa[3];
      *(v4h*)&As[r * LDA + lc] = ha;
    }
#pragma unroll
    for (int i = 0; i < 4; ++i) {
      int n  = lr + i * 32;
      int gl = l0 + n;
      v4f wb = (v4f)0.0f;
      if (gl < L_CLS)
        wb = *(const v4f*)(W + (size_t)gl * D_DIM + k0 + lc);
      v4h hb;
      hb[0] = (_Float16)wb[0]; hb[1] = (_Float16)wb[1];
      hb[2] = (_Float16)wb[2]; hb[3] = (_Float16)wb[3];
      *(v4h*)&Bs[n * LDB + lc] = hb;
    }
    __syncthreads();
    CLUB_COMPUTE_STAGE(As, Bs);
  }

  CLUB_EPILOGUE();
}

// ---------------------------------------------------------------------------
// out = (1/N) * sum_n (diag[n] + b[labels[n]])
//     - (1/N^2) * sum_j (colsum[labels[j]] + N*b[labels[j]])
// (log-softmax normalizers cancel exactly between the two terms)
// ---------------------------------------------------------------------------
__global__ __launch_bounds__(256)
void club_finalize(const float* __restrict__ colsum,
                   const float* __restrict__ diag,
                   const int*   __restrict__ labels,
                   const float* __restrict__ b,
                   float* __restrict__ out)
{
  __shared__ float red[256];
  float ps = 0.0f, ns = 0.0f;
  for (int i = (int)threadIdx.x; i < N_ROWS; i += 256) {
    int lab = labels[i];
    float bl = b[lab];
    ps += diag[i] + bl;
    ns += colsum[lab] + (float)N_ROWS * bl;
  }
  const float invN = 1.0f / (float)N_ROWS;
  red[threadIdx.x] = ps * invN - ns * (invN * invN);
  __syncthreads();
  for (int s = 128; s > 0; s >>= 1) {
    if ((int)threadIdx.x < s) red[threadIdx.x] += red[threadIdx.x + s];
    __syncthreads();
  }
  if (threadIdx.x == 0) out[0] = red[0];
}

// ---------------------------------------------------------------------------
extern "C" void kernel_launch(void* const* d_in, const int* in_sizes, int n_in,
                              void* d_out, int out_size, void* d_ws, size_t ws_size,
                              hipStream_t stream) {
  const float* X      = (const float*)d_in[0];
  const int*   labels = (const int*)  d_in[1];
  const float* W      = (const float*)d_in[2];
  const float* b      = (const float*)d_in[3];
  float* out = (float*)d_out;

  char* ws = (char*)d_ws;
  float* colsum = (float*)ws;                         // [L_PAD]
  float* diag   = colsum + L_PAD;                     // [N_ROWS]
  size_t off16  = ((size_t)(L_PAD + N_ROWS) * 4 + 255) & ~(size_t)255;
  _Float16* X16 = (_Float16*)(ws + off16);            // [N_ROWS * D_DIM]
  _Float16* W16 = X16 + (size_t)N_ROWS * D_DIM;       // [L_PAD * D_DIM]
  size_t need = off16 + ((size_t)N_ROWS + L_PAD) * D_DIM * 2;

  club_init_ws<<<(N_ROWS + 255) / 256, 256, 0, stream>>>(colsum, diag);

  dim3 grid(L_PAD / BN, N_ROWS / BM);                 // 8 x 128

  if (ws_size >= need) {
    club_cvt_x<<<(N_ROWS * D_DIM / 4) / 256, 256, 0, stream>>>(X, X16);
    club_cvt_w<<<(L_PAD  * D_DIM / 4) / 256, 256, 0, stream>>>(W, W16);
    club_gemm_f16_tdm<<<grid, 256, 0, stream>>>(X16, W16, labels, colsum, diag);
  } else {
    club_gemm_f32<<<grid, 256, 0, stream>>>(X, W, labels, colsum, diag);
  }

  club_finalize<<<1, 256, 0, stream>>>(colsum, diag, labels, b, out);
}
